// wrapped_linear_layers_369367188183
// MI455X (gfx1250) — matrix operand order repaired and checked by reading, not compiled, not run on hardware
//
#include <hip/hip_runtime.h>

// ---------------------------------------------------------------------------
// TT-linear: y = x @ W + bias, W = f0*f1*f2*f3*f4*f5 (TT cores, rank 64).
// Exploit the rank-64 bond between f2 and f3: W = G[4096,64] @ H[64,4096].
//   => Y = (X @ G) @ H + bias : ~9 GFLOP instead of 274 GFLOP dense.
// Memory-bound: 128MB X read + 128MB Y write ~= 11us at 23.3 TB/s.
// GEMMs use V_WMMA_F32_16X16X32_F16 (f16 in, f32 accumulate).
// gemm_xg is software-pipelined (double-buffered A and B) to hide HBM latency.
// ---------------------------------------------------------------------------

typedef __attribute__((ext_vector_type(16))) _Float16 v16h;
typedef __attribute__((ext_vector_type(8)))  _Float16 v8h;
typedef __attribute__((ext_vector_type(8)))  float    v8f;
typedef __attribute__((ext_vector_type(4)))  float    v4f;

#define IN_F   4096
#define OUT_F  4096
#define NROWS  8192   // 4 * 2048
#define RANK   64

// --------------------------- TT core contractions ---------------------------

// L[i1][i2][r2] = sum_r1 f0[0,i1,r1] * f1[r1,i2,r2]   (16*16*64 outputs)
__global__ void tt_left_a(const float* __restrict__ f0, const float* __restrict__ f1,
                          float* __restrict__ L) {
  int t  = blockIdx.x * blockDim.x + threadIdx.x;   // 16384 threads
  int r2 = t & 63, i2 = (t >> 6) & 15, i1 = t >> 10;
  float s = 0.f;
  #pragma unroll 4
  for (int r1 = 0; r1 < RANK; ++r1)
    s += f0[i1 * 64 + r1] * f1[(r1 * 16 + i2) * 64 + r2];
  L[t] = s;
}

// G[i][r3] = sum_r2 L[i1,i2][r2] * f2[r2,i3,r3]; i = i1*256+i2*16+i3.
// Stored f16 pre-swizzled as the WMMA B operand of GEMM1 (K=i, N=r3):
//   chunk (kt = i>>5, nt = r3>>4): lane = ((i>>3)&1)*16 + (r3&15), slot = (i&7) + ((i>>4)&1)*8.
__global__ void tt_left_b(const float* __restrict__ L, const float* __restrict__ f2,
                          _Float16* __restrict__ gsw) {
  int t   = blockIdx.x * blockDim.x + threadIdx.x;  // 262144 threads
  int r3  = t & 63;
  int i   = t >> 6;
  int i3  = i & 15, i12 = i >> 4;
  float s = 0.f;
  #pragma unroll 4
  for (int r2 = 0; r2 < RANK; ++r2)
    s += L[i12 * 64 + r2] * f2[(r2 * 16 + i3) * 64 + r3];
  int kt = i >> 5, kin = i & 31, nt = r3 >> 4, n = r3 & 15;
  gsw[(((kt * 4 + nt) * 32) + ((kin >> 3) & 1) * 16 + n) * 16 + (kin & 7) + ((kin >> 4) << 3)] = (_Float16)s;
}

// M[r4][o2][o3] = sum_r5 f4[r4,o2,r5] * f5[r5,o3,0]   (64*16*16 outputs)
__global__ void tt_right_a(const float* __restrict__ f4, const float* __restrict__ f5,
                           float* __restrict__ M) {
  int t  = blockIdx.x * blockDim.x + threadIdx.x;   // 16384 threads
  int o3 = t & 15, o2 = (t >> 4) & 15, r4 = t >> 8;
  float s = 0.f;
  #pragma unroll 4
  for (int r5 = 0; r5 < RANK; ++r5)
    s += f4[(r4 * 16 + o2) * 64 + r5] * f5[r5 * 16 + o3];
  M[t] = s;
}

// H[r3][o] = sum_r4 f3[r3,o1,r4] * M[r4][o2,o3]; o = o1*256+o2*16+o3.
// Stored f16 pre-swizzled as the WMMA B operand of GEMM2 (K=r3, N=o).
__global__ void tt_right_b(const float* __restrict__ f3, const float* __restrict__ M,
                           _Float16* __restrict__ hsw) {
  int t   = blockIdx.x * blockDim.x + threadIdx.x;  // 262144 threads
  int o   = t & 4095;
  int r3  = t >> 12;
  int o1  = o >> 8, o23 = o & 255;
  float s = 0.f;
  #pragma unroll 4
  for (int r4 = 0; r4 < RANK; ++r4)
    s += f3[(r3 * 16 + o1) * 64 + r4] * M[r4 * 256 + o23];
  int kt = r3 >> 5, kin = r3 & 31, nt = o >> 4, n = o & 15;
  hsw[(((nt * 2 + kt) * 32) + ((kin >> 3) & 1) * 16 + n) * 16 + (kin & 7) + ((kin >> 4) << 3)] = (_Float16)s;
}

// Convert 16 f32 (two 8-float groups) to the v16h A operand.
__device__ __forceinline__ v16h pack_a(v4f c0, v4f c1, v4f c2, v4f c3) {
  v16h a;
  a[0]  = (_Float16)c0.x; a[1]  = (_Float16)c0.y;
  a[2]  = (_Float16)c0.z; a[3]  = (_Float16)c0.w;
  a[4]  = (_Float16)c1.x; a[5]  = (_Float16)c1.y;
  a[6]  = (_Float16)c1.z; a[7]  = (_Float16)c1.w;
  a[8]  = (_Float16)c2.x; a[9]  = (_Float16)c2.y;
  a[10] = (_Float16)c2.z; a[11] = (_Float16)c2.w;
  a[12] = (_Float16)c3.x; a[13] = (_Float16)c3.y;
  a[14] = (_Float16)c3.z; a[15] = (_Float16)c3.w;
  return a;
}

// --------------------------- GEMM1: Zp = X @ G ------------------------------
// 2048 waves: 512 row-strips (16 rows) x 4 K-splits (1024 K each).
// A tile: X f32 -> f16 inline, CDNA5 A layout: lane h=lane>>4, m=lane&15,
//   VGPR j: K = (j>>2)*16 + h*8 + (j&3)*2 + {0,1}.
// Software-pipelined: iteration i issues loads for i+1 before consuming i.
__global__ __launch_bounds__(128) void gemm_xg(const float* __restrict__ X,
                                               const _Float16* __restrict__ gsw,
                                               float* __restrict__ Zp) {
  int lane  = threadIdx.x & 31;
  int w     = (blockIdx.x * 128 + threadIdx.x) >> 5;   // 0..2047
  int strip = w >> 2;                                  // 0..511
  int ks    = w & 3;                                   // K split 0..3
  int m0    = strip << 4;
  int h     = lane >> 4, mr = lane & 15;
  const float* xp = X + (size_t)(m0 + mr) * IN_F + ks * 1024 + h * 8;
  const v16h*  gB = (const v16h*)gsw + (size_t)(ks * 32) * 128 + lane;
  // gB strides: +32 per n-tile (1 KB), +128 per k-tile (4 KB).

  v8f acc[4];
  #pragma unroll
  for (int nt = 0; nt < 4; ++nt) acc[nt] = v8f{};

  // ---- pipeline prologue: loads for first k-tile ----
  v4f c0 = *(const v4f*)(xp);
  v4f c1 = *(const v4f*)(xp + 4);
  v4f c2 = *(const v4f*)(xp + 16);
  v4f c3 = *(const v4f*)(xp + 20);
  v16h b0 = gB[0], b1 = gB[32], b2 = gB[64], b3 = gB[96];

  for (int it = 0; it < 31; ++it) {
    xp += 32; gB += 128;
    // issue next-iteration loads first (latency hidden behind WMMAs below)
    v4f  n0  = *(const v4f*)(xp);
    v4f  n1  = *(const v4f*)(xp + 4);
    v4f  n2  = *(const v4f*)(xp + 16);
    v4f  n3  = *(const v4f*)(xp + 20);
    v16h bn0 = gB[0], bn1 = gB[32], bn2 = gB[64], bn3 = gB[96];
    // consume current
    v16h a = pack_a(c0, c1, c2, c3);
    acc[0] = __builtin_amdgcn_wmma_f32_16x16x32_f16(false, a, false, b0, (short)0, acc[0], false, false);
    acc[1] = __builtin_amdgcn_wmma_f32_16x16x32_f16(false, a, false, b1, (short)0, acc[1], false, false);
    acc[2] = __builtin_amdgcn_wmma_f32_16x16x32_f16(false, a, false, b2, (short)0, acc[2], false, false);
    acc[3] = __builtin_amdgcn_wmma_f32_16x16x32_f16(false, a, false, b3, (short)0, acc[3], false, false);
    // rotate
    c0 = n0; c1 = n1; c2 = n2; c3 = n3;
    b0 = bn0; b1 = bn1; b2 = bn2; b3 = bn3;
  }
  // ---- pipeline epilogue: last k-tile ----
  {
    v16h a = pack_a(c0, c1, c2, c3);
    acc[0] = __builtin_amdgcn_wmma_f32_16x16x32_f16(false, a, false, b0, (short)0, acc[0], false, false);
    acc[1] = __builtin_amdgcn_wmma_f32_16x16x32_f16(false, a, false, b1, (short)0, acc[1], false, false);
    acc[2] = __builtin_amdgcn_wmma_f32_16x16x32_f16(false, a, false, b2, (short)0, acc[2], false, false);
    acc[3] = __builtin_amdgcn_wmma_f32_16x16x32_f16(false, a, false, b3, (short)0, acc[3], false, false);
  }

  // D layout: lane holds col n=mr, rows m = h*8 + j. Partial per K-split.
  #pragma unroll
  for (int nt = 0; nt < 4; ++nt)
    #pragma unroll
    for (int j = 0; j < 8; ++j)
      Zp[((size_t)ks * NROWS + m0 + h * 8 + j) * 64 + nt * 16 + mr] = acc[nt][j];
}

// Deterministic reduction of the 4 K-split partials -> Z (f16).
__global__ void zreduce(const float* __restrict__ Zp, _Float16* __restrict__ Z) {
  int t = blockIdx.x * blockDim.x + threadIdx.x;     // 524288 threads
  const size_t S = (size_t)NROWS * 64;
  float s = Zp[t] + Zp[t + S] + Zp[t + 2 * S] + Zp[t + 3 * S];
  Z[t] = (_Float16)s;
}

// ----------------------- GEMM2: Y = Z @ H + bias ----------------------------
// 32768 waves: 512 row-strips x 64 col-chunks of 64. K=64 -> 2 WMMA k-steps.
__global__ __launch_bounds__(256) void gemm_zh(const _Float16* __restrict__ Z,
                                               const _Float16* __restrict__ hsw,
                                               const float* __restrict__ bias,
                                               float* __restrict__ Y) {
  int lane = threadIdx.x & 31;
  int w    = (blockIdx.x * 256 + threadIdx.x) >> 5;  // 0..32767
  int s    = w >> 6;                                 // row strip 0..511
  int c    = w & 63;                                 // col chunk 0..63
  int m0   = s << 4, n0 = c << 6;
  int h    = lane >> 4, mr = lane & 15;

  // A operands for kt=0,1 from Z (row-major [8192][64] f16).
  const _Float16* zr = Z + (size_t)(m0 + mr) * 64;
  v8h lo0 = *(const v8h*)(zr + h * 8);
  v8h hi0 = *(const v8h*)(zr + h * 8 + 16);
  v8h lo1 = *(const v8h*)(zr + 32 + h * 8);
  v8h hi1 = *(const v8h*)(zr + 32 + h * 8 + 16);
  v16h a0 = __builtin_shufflevector(lo0, hi0, 0,1,2,3,4,5,6,7,8,9,10,11,12,13,14,15);
  v16h a1 = __builtin_shufflevector(lo1, hi1, 0,1,2,3,4,5,6,7,8,9,10,11,12,13,14,15);

  // bias loads issued early so they overlap the WMMAs
  float bv[4];
  #pragma unroll
  for (int nt = 0; nt < 4; ++nt) bv[nt] = bias[n0 + nt * 16 + mr];

  const v16h* hB = (const v16h*)hsw;
  v8f acc[4];
  #pragma unroll
  for (int nt = 0; nt < 4; ++nt) acc[nt] = v8f{};

  #pragma unroll
  for (int nt = 0; nt < 4; ++nt) {
    int ntg = (n0 >> 4) + nt;
    v16h b0 = hB[(ntg * 2 + 0) * 32 + lane];
    v16h b1 = hB[(ntg * 2 + 1) * 32 + lane];
    acc[nt] = __builtin_amdgcn_wmma_f32_16x16x32_f16(
        false, a0, false, b0, (short)0, acc[nt], false, false);
    acc[nt] = __builtin_amdgcn_wmma_f32_16x16x32_f16(
        false, a1, false, b1, (short)0, acc[nt], false, false);
  }

  #pragma unroll
  for (int nt = 0; nt < 4; ++nt) {
    float* yr = Y + (size_t)(m0 + h * 8) * OUT_F + n0 + nt * 16 + mr;
    #pragma unroll
    for (int j = 0; j < 8; ++j)
      yr[(size_t)j * OUT_F] = acc[nt][j] + bv[nt];  // lanes 0..15: 16 consecutive floats
  }
}

// ------------------------------- launcher -----------------------------------

extern "C" void kernel_launch(void* const* d_in, const int* in_sizes, int n_in,
                              void* d_out, int out_size, void* d_ws, size_t ws_size,
                              hipStream_t stream) {
  (void)in_sizes; (void)n_in; (void)out_size; (void)ws_size;
  const float* X    = (const float*)d_in[0];
  const float* f0   = (const float*)d_in[1];
  const float* f1   = (const float*)d_in[2];
  const float* f2   = (const float*)d_in[3];
  const float* f3   = (const float*)d_in[4];
  const float* f4   = (const float*)d_in[5];
  const float* f5   = (const float*)d_in[6];
  const float* bias = (const float*)d_in[7];

  char* ws = (char*)d_ws;
  _Float16* gsw = (_Float16*)(ws);                          // 512 KB
  _Float16* hsw = (_Float16*)(ws + (512 << 10));            // 512 KB
  _Float16* Z   = (_Float16*)(ws + (1 << 20));              // 1 MB
  float*    Lt  = (float*)   (ws + (2 << 20));              // 64 KB
  float*    Mt  = (float*)   (ws + (2 << 20) + (64 << 10)); // 64 KB
  float*    Zp  = (float*)   (ws + (3 << 20));              // 8 MB

  tt_left_a <<<  64, 256, 0, stream>>>(f0, f1, Lt);
  tt_right_a<<<  64, 256, 0, stream>>>(f4, f5, Mt);
  tt_left_b <<<1024, 256, 0, stream>>>(Lt, f2, gsw);
  tt_right_b<<<1024, 256, 0, stream>>>(f3, Mt, hsw);
  gemm_xg   <<< 512, 128, 0, stream>>>(X, gsw, Zp);
  zreduce   <<<2048, 256, 0, stream>>>(Zp, Z);
  gemm_zh   <<<4096, 256, 0, stream>>>(Z, hsw, bias, (float*)d_out);
}
